// MultiAxisAttention_37520834297976
// MI455X (gfx1250) — matrix operand order repaired and checked by reading, not compiled, not run on hardware
//
#include <hip/hip_runtime.h>

// ---------------------------------------------------------------------------
// MultiAxisAttention for MI455X (gfx1250, wave32, WMMA bf16 + TDM)
// B=8, M=64, N=64, D=256, H=8
// ---------------------------------------------------------------------------

typedef __bf16 bf16;
typedef __attribute__((ext_vector_type(16))) __bf16 v16bf;
typedef __attribute__((ext_vector_type(8)))  __bf16 v8bf;
typedef __attribute__((ext_vector_type(8)))  float  v8f;

#define DD   256
#define MM   64
#define NN   64
#define BB   8
#define HH   8
#define RR   (BB * MM * NN)          // 32768 rows of flattened x

__device__ __forceinline__ v8f zero8() {
    v8f z = {0.f, 0.f, 0.f, 0.f, 0.f, 0.f, 0.f, 0.f};
    return z;
}

// A-fragment (16x32 bf16, M x K). Lane L<16: row L, K chunks {0..7,16..23};
// lanes 16..31: row L-16, K chunks {8..15,24..31}.  (ISA 7.12.2)
__device__ __forceinline__ v16bf load_frag_a(const bf16* base, int ld) {
    const int lane = threadIdx.x & 31;
    const bf16* p = base + (size_t)(lane & 15) * ld + ((lane >> 4) << 3);
    union { v16bf v; v8bf h[2]; } u;
    u.h[0] = *(const v8bf*)p;
    u.h[1] = *(const v8bf*)(p + 16);
    return u.v;
}

// B-fragment (32x16 bf16, K x N) read from a pre-transposed Bt[n][k] layout.
__device__ __forceinline__ v16bf load_frag_b(const bf16* bt_base, int ld) {
    const int lane = threadIdx.x & 31;
    const bf16* p = bt_base + (size_t)(lane & 15) * ld + ((lane >> 4) << 3);
    union { v16bf v; v8bf h[2]; } u;
    u.h[0] = *(const v8bf*)p;
    u.h[1] = *(const v8bf*)(p + 16);
    return u.v;
}

__device__ __forceinline__ v8f wmma_bf16(v16bf a, v16bf b, v8f c) {
    return __builtin_amdgcn_wmma_f32_16x16x32_bf16(
        false, a, false, b, (short)0, c, false, false);
}

// ---------------------------------------------------------------------------
// Tensor Data Mover: 2D tile (rows x cols bf16 elements, given row stride)
// D# layout per CDNA5 ISA 8.3/8.4; 6-arg (clang-23/therock) builtin form.
// ---------------------------------------------------------------------------
#if __has_builtin(__builtin_amdgcn_tensor_load_to_lds)
#define HAVE_TDM 1
typedef __attribute__((ext_vector_type(4))) unsigned int u32x4;
typedef __attribute__((ext_vector_type(8))) int          i32x8;
typedef __attribute__((ext_vector_type(4))) int          i32x4;

__device__ __forceinline__ void tdm_load_2d(const bf16* gsrc, unsigned lds_off,
                                            unsigned rows, unsigned cols,
                                            unsigned row_stride_elems) {
    unsigned long long ga = (unsigned long long)(uintptr_t)gsrc;
    u32x4 g0;
    g0[0] = 1u;                                            // count=1, user mode
    g0[1] = lds_off;                                       // lds_addr (bytes)
    g0[2] = (unsigned)(ga & 0xffffffffull);                // global_addr lo
    g0[3] = (unsigned)((ga >> 32) & 0x01ffffffull) | (2u << 30);  // addr hi | type=2
    i32x8 g1;
    g1[0] = (int)(1u << 16);                               // data_size=1 (2B)
    g1[1] = (int)((cols & 0xffffu) << 16);                 // tensor_dim0[15:0]
    g1[2] = (int)(((cols >> 16) & 0xffffu) | ((rows & 0xffffu) << 16)); // td0 hi|td1 lo
    g1[3] = (int)(((rows >> 16) & 0xffffu) | ((cols & 0xffffu) << 16)); // td1 hi|tile_dim0
    g1[4] = (int)(rows & 0xffffu);                         // tile_dim1 (tile_dim2=0)
    g1[5] = (int)row_stride_elems;                         // tensor_dim0_stride lo
    g1[6] = 0;                                             // stride hi
    g1[7] = 0;
    i32x4 z4 = {0, 0, 0, 0};
    i32x8 z8 = {0, 0, 0, 0, 0, 0, 0, 0};
    __builtin_amdgcn_tensor_load_to_lds(g0, g1, z4, z4, z8, 0);
}
#endif

// ---------------------------------------------------------------------------
// Prep kernels
// ---------------------------------------------------------------------------

__global__ void cvt_x_kernel(const float* __restrict__ in, bf16* __restrict__ out, int n) {
    int i = (blockIdx.x * blockDim.x + threadIdx.x) * 4;
    if (i + 3 < n) {
        float4 f = *(const float4*)(in + i);
        out[i + 0] = (bf16)f.x;
        out[i + 1] = (bf16)f.y;
        out[i + 2] = (bf16)f.z;
        out[i + 3] = (bf16)f.w;
    }
}

// dst[b][n][d] = src[b][d][n]
__global__ void tpose_cvt_kernel(const float* __restrict__ src, bf16* __restrict__ dst,
                                 int rows, int cols) {
    int bidx = blockIdx.y;
    int i = blockIdx.x * blockDim.x + threadIdx.x;
    if (i < rows * cols) {
        int d = i / cols, nn = i % cols;
        dst[(size_t)bidx * rows * cols + (size_t)nn * rows + d] =
            (bf16)src[(size_t)bidx * rows * cols + (size_t)d * cols + nn];
    }
}

// ot[v][h*256+d'] = o[h][d'][v]
__global__ void tpose_o_kernel(const float* __restrict__ o, bf16* __restrict__ ot) {
    int i = blockIdx.x * blockDim.x + threadIdx.x;
    int h = i >> 16;
    int rem = i & 65535;
    int dp = rem >> 8;
    int vv = rem & 255;
    ot[(size_t)vv * (HH * DD) + h * DD + dp] = (bf16)o[i];
}

// ---------------------------------------------------------------------------
// Stage A: Q[h]=x*q[h], K=x*k, V=x*v. grid=(RR/128, 10), block=256.
// Weight matrix (128KB) staged once into LDS; B-frags become ds_load_b128.
// ---------------------------------------------------------------------------
__global__ void __launch_bounds__(256) gemm_xw_kernel(
    const bf16* __restrict__ xb, const bf16* __restrict__ wt,
    bf16* __restrict__ Qb, bf16* __restrict__ Kb, bf16* __restrict__ Vb) {

    __shared__ bf16 Wl[DD * DD];                 // 128 KB
    const int mat = blockIdx.y;
    const bf16* wmat = wt + (size_t)mat * (DD * DD);

    for (int i = threadIdx.x * 8; i < DD * DD; i += 256 * 8)
        *(v8bf*)&Wl[i] = *(const v8bf*)(wmat + i);
    __syncthreads();

    const int wave = threadIdx.x >> 5;
    const int lane = threadIdx.x & 31;
    const int r0   = blockIdx.x * 128 + wave * 16;
    const bf16* arow = xb + (size_t)r0 * DD;

    v8f acc[16];
#pragma unroll
    for (int t = 0; t < 16; t++) acc[t] = zero8();

    for (int k0 = 0; k0 < DD; k0 += 32) {
        if (k0 + 32 < DD)
            __builtin_prefetch(arow + (size_t)(lane & 15) * DD + k0 + 32, 0, 0);
        v16bf aF = load_frag_a(arow + k0, DD);
#pragma unroll
        for (int half = 0; half < 2; half++) {
            v16bf bF[8];
#pragma unroll
            for (int j = 0; j < 8; j++)
                bF[j] = load_frag_b(&Wl[(half * 8 + j) * 16 * DD + k0], DD);
#pragma unroll
            for (int j = 0; j < 8; j++)
                acc[half * 8 + j] = wmma_bf16(aF, bF[j], acc[half * 8 + j]);
        }
    }

    const int b = r0 >> 12;
    const int t = r0 & 4095;
    bf16* outp;
    size_t base;
    if (mat < 8)       { base = ((size_t)(b * HH + mat) * (MM * NN) + t) * DD; outp = Qb; }
    else if (mat == 8) { base = (size_t)r0 * DD; outp = Kb; }
    else               { base = (size_t)r0 * DD; outp = Vb; }

    const int rlo = (lane >> 4) * 8;
    const int c   = lane & 15;
#pragma unroll
    for (int nt = 0; nt < 16; nt++)
#pragma unroll
        for (int i = 0; i < 8; i++)
            outp[base + (size_t)(rlo + i) * DD + nt * 16 + c] = (bf16)acc[nt][i];
}

// ---------------------------------------------------------------------------
// Stage B: attention. One block per (b, h, slice s). Q/K tiles staged by TDM.
// ---------------------------------------------------------------------------
__global__ void __launch_bounds__(256) attn_kernel(
    const bf16* __restrict__ Qb, const bf16* __restrict__ Kb,
    const bf16* __restrict__ Vb, bf16* __restrict__ Ob) {

    __shared__ bf16  Qs[64][DD];      // 32 KB
    __shared__ bf16  Ks[64][DD];      // 32 KB
    __shared__ bf16  Vt[DD][80];      // 40 KB  (V^T, 160B pitch -> 32B aligned rows)
    __shared__ float Sl[64][68];      // ~17 KB
    __shared__ bf16  Pl[64][64];      // 8 KB

    const int s = blockIdx.x;
    const int h = blockIdx.y;
    const int b = blockIdx.z;

    size_t qbase, kbase;
    int qstride, kstride;
    if (h < 4) {
        qbase = ((size_t)((b * HH + h) * (MM * NN)) + s) * DD; qstride = NN * DD;
        kbase = ((size_t)(b * (MM * NN)) + s) * DD;            kstride = NN * DD;
    } else {
        qbase = ((size_t)((b * HH + h) * (MM * NN) + s * NN)) * DD; qstride = DD;
        kbase = ((size_t)(b * (MM * NN) + s * NN)) * DD;            kstride = DD;
    }

    // ---- V: manual transposed staging (all threads) ----
    {
        const int row = threadIdx.x >> 2;
        const int c0  = (threadIdx.x & 3) * 64;
        const bf16* vsrc = Vb + kbase + (size_t)row * kstride + c0;
#pragma unroll
        for (int j = 0; j < 8; j++) {
            v8bf vv = *(const v8bf*)(vsrc + j * 8);
#pragma unroll
            for (int e = 0; e < 8; e++) Vt[c0 + j * 8 + e][row] = vv[e];
        }
    }

#ifdef HAVE_TDM
    // ---- Q, K: Tensor Data Mover (wave 0 issues 2 DMAs, fences TENSORcnt) ----
    if (threadIdx.x < 32) {
        tdm_load_2d(Qb + qbase, (unsigned)(uintptr_t)&Qs[0][0], 64, DD, (unsigned)qstride);
        tdm_load_2d(Kb + kbase, (unsigned)(uintptr_t)&Ks[0][0], 64, DD, (unsigned)kstride);
        __builtin_amdgcn_s_wait_tensorcnt(0);
    }
#else
    {
        const int row = threadIdx.x >> 2;
        const int c0  = (threadIdx.x & 3) * 64;
        const bf16* qsrc = Qb + qbase + (size_t)row * qstride + c0;
        const bf16* ksrc = Kb + kbase + (size_t)row * kstride + c0;
#pragma unroll
        for (int j = 0; j < 8; j++) {
            *(v8bf*)&Qs[row][c0 + j * 8] = *(const v8bf*)(qsrc + j * 8);
            *(v8bf*)&Ks[row][c0 + j * 8] = *(const v8bf*)(ksrc + j * 8);
        }
    }
#endif
    __syncthreads();

    const int wave = threadIdx.x >> 5;
    const int lane = threadIdx.x & 31;

    // ---- S = Q * K^T (64x64, K=256); wave: 16 rows x 32 cols ----
    {
        const int m0 = (wave & 3) * 16;
        const int n0 = (wave >> 2) * 32;
        v8f acc0 = zero8(), acc1 = zero8();
        for (int k0 = 0; k0 < DD; k0 += 32) {
            v16bf aF = load_frag_a(&Qs[m0][k0], DD);
            v16bf b0 = load_frag_b(&Ks[n0][k0], DD);
            v16bf b1 = load_frag_b(&Ks[n0 + 16][k0], DD);
            acc0 = wmma_bf16(aF, b0, acc0);
            acc1 = wmma_bf16(aF, b1, acc1);
        }
        const int rlo = (lane >> 4) * 8, c = lane & 15;
#pragma unroll
        for (int i = 0; i < 8; i++) {
            Sl[m0 + rlo + i][n0 + c]      = acc0[i];
            Sl[m0 + rlo + i][n0 + 16 + c] = acc1[i];
        }
    }
    __syncthreads();

    // ---- row softmax ----
    if (threadIdx.x < 64) {
        const int r = threadIdx.x;
        float mx = -3.4e38f;
        for (int z = 0; z < 64; z++) mx = fmaxf(mx, Sl[r][z]);
        float sum = 0.f;
        for (int z = 0; z < 64; z++) { float e = __expf(Sl[r][z] - mx); Sl[r][z] = e; sum += e; }
        float inv = 1.f / sum;
        for (int z = 0; z < 64; z++) Pl[r][z] = (bf16)(Sl[r][z] * inv);
    }
    __syncthreads();

    // ---- O = P * V (64x256, K=64); wave: 16 rows x 128 cols ----
    {
        const int m0 = (wave & 3) * 16;
        const int n0 = (wave >> 2) * 128;
        v8f acc[8];
#pragma unroll
        for (int j = 0; j < 8; j++) acc[j] = zero8();
        for (int k0 = 0; k0 < 64; k0 += 32) {
            v16bf aF = load_frag_a(&Pl[m0][k0], 64);
            v16bf bF[8];
#pragma unroll
            for (int j = 0; j < 8; j++)
                bF[j] = load_frag_b(&Vt[n0 + j * 16][k0], 80);
#pragma unroll
            for (int j = 0; j < 8; j++)
                acc[j] = wmma_bf16(aF, bF[j], acc[j]);
        }
        const int rlo = (lane >> 4) * 8, c = lane & 15;
#pragma unroll
        for (int j = 0; j < 8; j++)
#pragma unroll
            for (int i = 0; i < 8; i++)
                Ob[qbase + (size_t)(m0 + rlo + i) * qstride + n0 + j * 16 + c] =
                    (bf16)acc[j][i];
    }
}

// ---------------------------------------------------------------------------
// Stage C: Z = sum_h O[h]*o[h], fused K=2048. Per-head 128KB B-slab in LDS.
// ---------------------------------------------------------------------------
__global__ void __launch_bounds__(256) gemm_out_kernel(
    const bf16* __restrict__ Ob, const bf16* __restrict__ ot,
    float* __restrict__ Z) {

    __shared__ bf16 Ol[DD * DD];                 // 128 KB per-head slab [n][k]
    const int wave = threadIdx.x >> 5;
    const int lane = threadIdx.x & 31;
    const int r0 = blockIdx.x * 128 + wave * 16;
    const int b = r0 >> 12;
    const int t = r0 & 4095;

    v8f acc[16];
#pragma unroll
    for (int n = 0; n < 16; n++) acc[n] = zero8();

    for (int h = 0; h < HH; h++) {
        __syncthreads();
        for (int i = threadIdx.x * 8; i < DD * DD; i += 256 * 8) {
            int n  = i >> 8;
            int kk = i & 255;
            *(v8bf*)&Ol[i] = *(const v8bf*)(ot + (size_t)n * (HH * DD) + h * DD + kk);
        }
        __syncthreads();

        const bf16* abase = Ob + ((size_t)(b * HH + h) * (MM * NN) + t) * DD;
        for (int koff = 0; koff < DD; koff += 32) {
            if (koff + 32 < DD)
                __builtin_prefetch(abase + (size_t)(lane & 15) * DD + koff + 32, 0, 0);
            v16bf aF = load_frag_a(abase + koff, DD);
#pragma unroll
            for (int half = 0; half < 2; half++) {
                v16bf bF[8];
#pragma unroll
                for (int j = 0; j < 8; j++)
                    bF[j] = load_frag_b(&Ol[(half * 8 + j) * 16 * DD + koff], DD);
#pragma unroll
                for (int j = 0; j < 8; j++)
                    acc[half * 8 + j] = wmma_bf16(aF, bF[j], acc[half * 8 + j]);
            }
        }
    }

    const int rlo = (lane >> 4) * 8, c = lane & 15;
#pragma unroll
    for (int nt = 0; nt < 16; nt++)
#pragma unroll
        for (int i = 0; i < 8; i++)
            Z[(size_t)(r0 + rlo + i) * DD + nt * 16 + c] = acc[nt][i];
}

// ---------------------------------------------------------------------------
// Host launcher
// ---------------------------------------------------------------------------
extern "C" void kernel_launch(void* const* d_in, const int* in_sizes, int n_in,
                              void* d_out, int out_size, void* d_ws, size_t ws_size,
                              hipStream_t stream) {
    const float* x = (const float*)d_in[0];
    const float* q = (const float*)d_in[1];
    const float* k = (const float*)d_in[2];
    const float* v = (const float*)d_in[3];
    const float* o = (const float*)d_in[4];
    float* Z = (float*)d_out;

    bf16* xb  = (bf16*)d_ws;
    bf16* wt  = xb + (size_t)RR * DD;
    bf16* ot  = wt + (size_t)10 * DD * DD;
    bf16* Kb  = ot + (size_t)DD * HH * DD;
    bf16* Vb  = Kb + (size_t)RR * DD;
    bf16* Qb  = Vb + (size_t)RR * DD;
    bf16* Obf = Qb + (size_t)HH * RR * DD;

    cvt_x_kernel<<<(RR * DD) / (256 * 4), 256, 0, stream>>>(x, xb, RR * DD);
    tpose_cvt_kernel<<<dim3((DD * DD) / 256, 8), 256, 0, stream>>>(q, wt, DD, DD);
    tpose_cvt_kernel<<<dim3((DD * DD) / 256, 1), 256, 0, stream>>>(k, wt + (size_t)8 * DD * DD, DD, DD);
    tpose_cvt_kernel<<<dim3((DD * DD) / 256, 1), 256, 0, stream>>>(v, wt + (size_t)9 * DD * DD, DD, DD);
    tpose_o_kernel<<<(HH * DD * DD) / 256, 256, 0, stream>>>(o, ot);

    gemm_xw_kernel<<<dim3(RR / 128, 10), 256, 0, stream>>>(xb, wt, Qb, Kb, Vb);
    attn_kernel<<<dim3(64, HH, BB), 256, 0, stream>>>(Qb, Kb, Vb, Obf);
    gemm_out_kernel<<<RR / 128, 256, 0, stream>>>(Obf, ot, Z);
}
